// JetBlock_45268955300516
// MI455X (gfx1250) — matrix operand order repaired and checked by reading, not compiled, not run on hardware
//
#include <hip/hip_runtime.h>
#include <hip/hip_bf16.h>
#include <stdint.h>

// ---------------------------------------------------------------------------
// MI455X (gfx1250) transformer block.
// - All GEMM-shaped math: V_WMMA_F32_16X16X32_BF16 (wave32 fragments, ISA 7.12.2)
// - All LDS staging: GLOBAL_LOAD_ASYNC_TO_LDS_B128 (ASYNCcnt) with counted
//   s_wait_asynccnt double buffering (ISA ch.10.4 / 15.18.3)
// - Weights pre-transposed to [N][K] bf16 so A and B tiles are both straight
//   row-major copies (no per-step LDS transpose scatter).
// ---------------------------------------------------------------------------

typedef unsigned short u16;
typedef __attribute__((ext_vector_type(16))) __bf16 v16bf;
typedef __attribute__((ext_vector_type(8)))  float  v8f;

union Frag16 { uint4 q[2]; v16bf v; };

__device__ __forceinline__ u16 f2bf(float f) {
  unsigned int u = __float_as_uint(f);
  u += 0x7FFFu + ((u >> 16) & 1u);          // round-to-nearest-even
  return (u16)(u >> 16);
}

// async copy 16B per lane: global -> LDS (ASYNCcnt-tracked)
__device__ __forceinline__ void async_b128(void* lds, const u16* g) {
  unsigned int l = (unsigned int)(uintptr_t)lds;        // flat[31:0] == LDS addr
  unsigned long long ga = (unsigned long long)(uintptr_t)g;
  asm volatile("global_load_async_to_lds_b128 %0, %1, off"
               :: "v"(l), "v"(ga) : "memory");
}
__device__ __forceinline__ void wait_async0() {
  asm volatile("s_wait_asynccnt 0x0" ::: "memory");
}
__device__ __forceinline__ void wait_async8() {        // <=8 outstanding
  asm volatile("s_wait_asynccnt 0x8" ::: "memory");
}

// problem constants
#define Bb     2
#define Tt     2048
#define Cc     2048
#define Hh     16
#define DHd    128
#define INNERn 8192
#define ROWS   (Bb * Tt)                    // 4096

// ===========================================================================
// fp32 [K][N] -> bf16 [N][K] tiled transpose (one-time weight prep)
// ===========================================================================
__global__ __launch_bounds__(256) void transpose_cvt_kernel(
    const float* __restrict__ in, u16* __restrict__ out, int K, int N) {
  __shared__ float tile[32][33];
  int k0 = blockIdx.y * 32, n0 = blockIdx.x * 32;
  int tx = threadIdx.x & 31, ty = threadIdx.x >> 5;     // 32 x 8
  #pragma unroll
  for (int i = ty; i < 32; i += 8)
    tile[i][tx] = in[(size_t)(k0 + i) * N + n0 + tx];
  __syncthreads();
  #pragma unroll
  for (int i = ty; i < 32; i += 8)
    out[(size_t)(n0 + i) * K + k0 + tx] = f2bf(tile[tx][i]);
}

// ===========================================================================
// LayerNorm over C=2048, one block per row, writes bf16
// ===========================================================================
__global__ __launch_bounds__(256) void ln_bf16_kernel(
    const float* __restrict__ X, const float* __restrict__ w,
    const float* __restrict__ bta, u16* __restrict__ out) {
  __shared__ float s1[256], s2[256];
  int row = blockIdx.x, tid = threadIdx.x;
  const float* xr = X + (size_t)row * Cc;
  float sum = 0.f, sq = 0.f;
  for (int c = tid; c < Cc; c += 256) { float v = xr[c]; sum += v; sq += v * v; }
  s1[tid] = sum; s2[tid] = sq; __syncthreads();
  for (int off = 128; off > 0; off >>= 1) {
    if (tid < off) { s1[tid] += s1[tid + off]; s2[tid] += s2[tid + off]; }
    __syncthreads();
  }
  float mu = s1[0] * (1.0f / Cc);
  float var = s2[0] * (1.0f / Cc) - mu * mu;
  float rstd = rsqrtf(var + 1e-5f);
  for (int c = tid; c < Cc; c += 256)
    out[(size_t)row * Cc + c] = f2bf((xr[c] - mu) * rstd * w[c] + bta[c]);
}

// ===========================================================================
// WMMA GEMM: C[M,N] (f32, += R if HASR) = A[M,K](bf16) * Bt[N,K](bf16)
// Tile 128x128x64, 256 threads = 8 waves (2x4), wave tile 64x32.
// Double-buffered async-to-LDS staging; fragment loads are 2x ds b128.
// ===========================================================================
#define GBM 128
#define GBN 128
#define GBK 64

template <bool HASR>
__global__ __launch_bounds__(256) void gemm_bf16_kernel(
    const u16* __restrict__ A, const u16* __restrict__ Bt,
    float* __restrict__ C, const float* __restrict__ R,
    int M, int N, int K) {
  __shared__ __align__(16) u16 As[2][GBM][GBK];
  __shared__ __align__(16) u16 Bs[2][GBN][GBK];

  const int tid = threadIdx.x;
  const int lane = tid & 31, half = lane >> 4, l16 = lane & 15;
  const int wv = tid >> 5, wr = wv >> 2, wc = wv & 3;   // 2 x 4 wave grid
  const int m0 = blockIdx.y * GBM, n0 = blockIdx.x * GBN;

  // per-thread staging: 4 b128 chunks for A, 4 for B  (8 asyncs / wave / buf)
  auto stage = [&](int bufi, int k0) {
    #pragma unroll
    for (int it = 0; it < 4; ++it) {
      int v = tid + it * 256;                 // 0..1023 vec8 chunks
      int row = v >> 3, koff = (v & 7) << 3;  // 8 chunks per 64-wide row
      async_b128(&As[bufi][row][koff], A  + (size_t)(m0 + row) * K + k0 + koff);
      async_b128(&Bs[bufi][row][koff], Bt + (size_t)(n0 + row) * K + k0 + koff);
    }
  };

  v8f acc[4][2];
  #pragma unroll
  for (int i = 0; i < 4; ++i)
    #pragma unroll
    for (int j = 0; j < 2; ++j)
      for (int e = 0; e < 8; ++e) acc[i][j][e] = 0.f;

  stage(0, 0);
  int buf = 0;
  for (int k0 = 0; k0 < K; k0 += GBK) {
    if (k0 + GBK < K) {
      stage(buf ^ 1, k0 + GBK);   // prefetch next tile while current computes
      wait_async8();              // older buffer's 8 issues retired
    } else {
      wait_async0();
    }
    __syncthreads();              // current buffer resident for all waves

    #pragma unroll
    for (int c = 0; c < 2; ++c) {                 // two K=32 wmma steps
      Frag16 af[4], bfr[2];
      #pragma unroll
      for (int i = 0; i < 4; ++i) {
        // A 16x32 frag: lanes<16 K=0..7,16..23 ; lanes>=16 K=8..15,24..31
        const u16* p = &As[buf][wr * 64 + i * 16 + l16][c * 32 + half * 8];
        af[i].q[0] = *(const uint4*)p;
        af[i].q[1] = *(const uint4*)(p + 16);
      }
      #pragma unroll
      for (int j = 0; j < 2; ++j) {
        // B 32x16 frag: lanes<16 K=0..7,16..23 ; lanes>=16 K=8..15,24..31 (N = lane&15)
        const u16* p = &Bs[buf][wc * 32 + j * 16 + l16][c * 32 + half * 8];
        bfr[j].q[0] = *(const uint4*)p;
        bfr[j].q[1] = *(const uint4*)(p + 16);
      }
      #pragma unroll
      for (int i = 0; i < 4; ++i)
        #pragma unroll
        for (int j = 0; j < 2; ++j)
          acc[i][j] = __builtin_amdgcn_wmma_f32_16x16x32_bf16(
              false, af[i].v, false, bfr[j].v, (short)0, acc[i][j], false, false);
    }
    __syncthreads();              // all waves done reading buf before overwrite
    buf ^= 1;
  }

  // epilogue: VGPR r holds M=r (lanes 0-15) / M=r+8 (lanes 16-31)
  #pragma unroll
  for (int i = 0; i < 4; ++i)
    #pragma unroll
    for (int j = 0; j < 2; ++j)
      #pragma unroll
      for (int r = 0; r < 8; ++r) {
        int row = m0 + wr * 64 + i * 16 + r + half * 8;
        int col = n0 + wc * 32 + j * 16 + l16;
        size_t idx = (size_t)row * N + col;
        float vlt = acc[i][j][r];
        if constexpr (HASR) vlt += R[idx];
        C[idx] = vlt;
      }
}

// ===========================================================================
// RoPE + qkv split: qkv f32 [B*T, 3C] -> q,k bf16 [B,H,T,DH];
// v written pre-transposed: [B,H,DH,T] (so flash V staging is a copy)
// ===========================================================================
__global__ __launch_bounds__(64) void rope_split_kernel(
    const float* __restrict__ qkv, const float* __restrict__ cosT,
    const float* __restrict__ sinT, u16* __restrict__ Q,
    u16* __restrict__ Ko, u16* __restrict__ Vt) {
  int idx = blockIdx.x;
  int h = idx % Hh;
  int t = (idx / Hh) % Tt;
  int b = idx / (Hh * Tt);
  int i = threadIdx.x;                        // rotary pair 0..63
  const float* base = qkv + (size_t)(b * Tt + t) * (3 * Cc);
  float cv = cosT[t * 64 + i], sv = sinT[t * 64 + i];
  int bh = b * Hh + h;
  size_t orow = ((size_t)bh * Tt + t) * DHd;

  float x1 = base[h * DHd + 2 * i], x2 = base[h * DHd + 2 * i + 1];
  Q[orow + 2 * i]     = f2bf(x1 * cv - x2 * sv);
  Q[orow + 2 * i + 1] = f2bf(x1 * sv + x2 * cv);

  x1 = base[Cc + h * DHd + 2 * i]; x2 = base[Cc + h * DHd + 2 * i + 1];
  Ko[orow + 2 * i]     = f2bf(x1 * cv - x2 * sv);
  Ko[orow + 2 * i + 1] = f2bf(x1 * sv + x2 * cv);

  Vt[((size_t)bh * DHd + 2 * i)     * Tt + t] = f2bf(base[2 * Cc + h * DHd + 2 * i]);
  Vt[((size_t)bh * DHd + 2 * i + 1) * Tt + t] = f2bf(base[2 * Cc + h * DHd + 2 * i + 1]);
}

// ===========================================================================
// WMMA flash attention (causal). 128 threads = 4 waves, 64 queries / block.
// K and V(pre-transposed) tiles staged via async-to-LDS; S = q@K^T (8 wmma),
// online softmax via shfl_xor in 16-lane halves, P restaged through LDS into
// the A-fragment layout, O += P@V (8 wmma).
// ===========================================================================
#define FA_BM 64
#define FA_BN 32

__global__ __launch_bounds__(128) void flash_attn_kernel(
    const u16* __restrict__ Q, const u16* __restrict__ Km,
    const u16* __restrict__ VtG, u16* __restrict__ Y) {
  const int bh = blockIdx.x;                  // 0..B*H-1
  const int b = bh / Hh, h = bh % Hh;
  const int qbase = blockIdx.y * FA_BM;
  const int tid = threadIdx.x;
  const int wv = tid >> 5, lane = tid & 31, half = lane >> 4, l16 = lane & 15;

  __shared__ __align__(16) u16 Ks[FA_BN][DHd];     // [key][dh] == Bt for q@K^T
  __shared__ __align__(16) u16 Vs[DHd][FA_BN];     // [dh][key] == Bt for P@V
  __shared__ __align__(16) u16 Ps[4][16][FA_BN];   // per-wave P staging

  const u16* Qb = Q + ((size_t)bh * Tt + qbase) * DHd;
  const u16* Kb = Km + (size_t)bh * Tt * DHd;
  const u16* Vb = VtG + (size_t)bh * DHd * Tt;

  // resident q fragments: 16x128 per wave = 4 chunks of 16x32 (A layout)
  Frag16 qf[4];
  {
    int qrow = wv * 16 + l16;
    #pragma unroll
    for (int c = 0; c < 4; ++c) {
      const u16* p = Qb + (size_t)qrow * DHd + c * 32 + half * 8;
      qf[c].q[0] = *(const uint4*)p;
      qf[c].q[1] = *(const uint4*)(p + 16);
    }
  }

  v8f Oacc[8];
  #pragma unroll
  for (int j = 0; j < 8; ++j)
    for (int e = 0; e < 8; ++e) Oacc[j][e] = 0.f;
  float m_r[8], l_r[8];
  #pragma unroll
  for (int r = 0; r < 8; ++r) { m_r[r] = -1.0e30f; l_r[r] = 0.f; }

  const int ktiles = (qbase + FA_BM - 1) / FA_BN + 1;   // causal bound
  for (int kt = 0; kt < ktiles; ++kt) {
    __syncthreads();                       // previous iteration reads done
    // K tile 32x128 copy; V tile 128x32 copy (already transposed in global)
    #pragma unroll
    for (int it = 0; it < 4; ++it) {
      int v = tid + it * 128;              // K: 0..511 vec8 chunks
      int row = v >> 4, coff = (v & 15) << 3;
      async_b128(&Ks[row][coff], Kb + ((size_t)kt * FA_BN + row) * DHd + coff);
      int dh = v >> 2, kc = (v & 3) << 3;  // V: 128 rows x 4 chunks
      async_b128(&Vs[dh][kc], Vb + (size_t)dh * Tt + kt * FA_BN + kc);
    }
    wait_async0();
    __syncthreads();

    // S = q @ K^T  (16x32 per wave, 2 col tiles x 4 dh-chunks)
    v8f Sacc[2];
    #pragma unroll
    for (int ct = 0; ct < 2; ++ct)
      for (int e = 0; e < 8; ++e) Sacc[ct][e] = 0.f;
    #pragma unroll
    for (int c = 0; c < 4; ++c)
      #pragma unroll
      for (int ct = 0; ct < 2; ++ct) {
        Frag16 bf;
        const u16* p = &Ks[ct * 16 + l16][c * 32 + half * 8];
        bf.q[0] = *(const uint4*)p;
        bf.q[1] = *(const uint4*)(p + 16);
        Sacc[ct] = __builtin_amdgcn_wmma_f32_16x16x32_bf16(
            false, qf[c].v, false, bf.v, (short)0, Sacc[ct], false, false);
      }

    // scale + causal mask
    float pvals[2][8];
    #pragma unroll
    for (int ct = 0; ct < 2; ++ct)
      #pragma unroll
      for (int r = 0; r < 8; ++r) {
        int qg = qbase + wv * 16 + r + half * 8;
        int kg = kt * FA_BN + ct * 16 + l16;
        float s = Sacc[ct][r] * 0.08838834764831845f;   // 1/sqrt(128)
        pvals[ct][r] = (kg <= qg) ? s : -3.0e38f;
      }

    // online softmax per row (rows of a half live in that half's lanes)
    #pragma unroll
    for (int r = 0; r < 8; ++r) {
      float t = fmaxf(pvals[0][r], pvals[1][r]);
      for (int off = 8; off >= 1; off >>= 1)
        t = fmaxf(t, __shfl_xor(t, off, 32));
      float mnew = fmaxf(m_r[r], t);
      float alpha = __expf(m_r[r] - mnew);
      m_r[r] = mnew;
      float rs = 0.f;
      #pragma unroll
      for (int ct = 0; ct < 2; ++ct) {
        float p = (pvals[ct][r] > -1.0e38f) ? __expf(pvals[ct][r] - mnew) : 0.f;
        pvals[ct][r] = p;
        rs += p;
      }
      for (int off = 8; off >= 1; off >>= 1) rs += __shfl_xor(rs, off, 32);
      l_r[r] = l_r[r] * alpha + rs;
      #pragma unroll
      for (int j = 0; j < 8; ++j) Oacc[j][r] *= alpha;   // rescale O rows
      #pragma unroll
      for (int ct = 0; ct < 2; ++ct)
        Ps[wv][r + half * 8][ct * 16 + l16] = f2bf(pvals[ct][r]);
    }

    // O += P @ V   (A frag from Ps, B frags from Vs; one K=32 step)
    Frag16 pf;
    {
      const u16* pp = &Ps[wv][l16][half * 8];
      pf.q[0] = *(const uint4*)pp;
      pf.q[1] = *(const uint4*)(pp + 16);
    }
    #pragma unroll
    for (int j = 0; j < 8; ++j) {
      Frag16 vf;
      const u16* pv2 = &Vs[j * 16 + l16][half * 8];
      vf.q[0] = *(const uint4*)pv2;
      vf.q[1] = *(const uint4*)(pv2 + 16);
      Oacc[j] = __builtin_amdgcn_wmma_f32_16x16x32_bf16(
          false, pf.v, false, vf.v, (short)0, Oacc[j], false, false);
    }
  }

  // write y as bf16 in [B,T,C] (ready as GEMM A operand for W_o)
  #pragma unroll
  for (int j = 0; j < 8; ++j)
    #pragma unroll
    for (int r = 0; r < 8; ++r) {
      int tq = qbase + wv * 16 + r + half * 8;
      int col = h * DHd + j * 16 + l16;
      float val = Oacc[j][r] / l_r[r];
      Y[(size_t)(b * Tt + tq) * Cc + col] = f2bf(val);
    }
}

// ===========================================================================
// SwiGLU elementwise: m = silu(g1) * g2 -> bf16
// ===========================================================================
__global__ void swiglu_kernel(const float* __restrict__ g1,
                              const float* __restrict__ g2,
                              u16* __restrict__ out, long n) {
  long i = (long)blockIdx.x * blockDim.x + threadIdx.x;
  long stride = (long)gridDim.x * blockDim.x;
  for (; i < n; i += stride) {
    float a = g1[i];
    float s = a / (1.f + __expf(-a));
    out[i] = f2bf(s * g2[i]);
  }
}

// ===========================================================================
// Orchestration
// ===========================================================================
extern "C" void kernel_launch(void* const* d_in, const int* in_sizes, int n_in,
                              void* d_out, int out_size, void* d_ws,
                              size_t ws_size, hipStream_t stream) {
  (void)in_sizes; (void)n_in; (void)out_size; (void)ws_size;

  const float* x    = (const float*)d_in[0];
  const float* cosT = (const float*)d_in[1];
  const float* sinT = (const float*)d_in[2];
  const float* ln1w = (const float*)d_in[3];
  const float* ln1b = (const float*)d_in[4];
  const float* ln2w = (const float*)d_in[5];
  const float* ln2b = (const float*)d_in[6];
  const float* Wqkv = (const float*)d_in[7];
  const float* Wo   = (const float*)d_in[8];
  const float* W1   = (const float*)d_in[9];
  const float* W2   = (const float*)d_in[10];
  const float* W3   = (const float*)d_in[11];
  float* out = (float*)d_out;

  char* wsp = (char*)d_ws;
  size_t off = 0;
  auto alloc = [&](size_t bytes) -> void* {
    void* p = wsp + off;
    off += (bytes + 255) & ~(size_t)255;
    return p;
  };

  u16*   wqkv_t = (u16*)  alloc((size_t)Cc * 3 * Cc * 2);   // [3C][C]
  u16*   wo_t   = (u16*)  alloc((size_t)Cc * Cc * 2);       // [C][C]
  u16*   w1_t   = (u16*)  alloc((size_t)Cc * INNERn * 2);   // [INNER][C]
  u16*   w2_t   = (u16*)  alloc((size_t)Cc * INNERn * 2);   // [INNER][C]
  u16*   w3_t   = (u16*)  alloc((size_t)INNERn * Cc * 2);   // [C][INNER]
  u16*   a_bf   = (u16*)  alloc((size_t)ROWS * Cc * 2);
  float* qkvf   = (float*)alloc((size_t)ROWS * 3 * Cc * 4);
  u16*   qb     = (u16*)  alloc((size_t)ROWS * Cc * 2);
  u16*   kb     = (u16*)  alloc((size_t)ROWS * Cc * 2);
  u16*   vtb    = (u16*)  alloc((size_t)ROWS * Cc * 2);     // [B,H,DH,T]
  u16*   yb     = (u16*)  alloc((size_t)ROWS * Cc * 2);
  float* x2     = (float*)alloc((size_t)ROWS * Cc * 4);
  u16*   hb     = (u16*)  alloc((size_t)ROWS * Cc * 2);
  float* g1     = (float*)alloc((size_t)ROWS * INNERn * 4);
  float* g2     = (float*)alloc((size_t)ROWS * INNERn * 4);
  u16*   mb     = (u16*)  alloc((size_t)ROWS * INNERn * 2);

  // one-time weight transpose+convert: [K][N] f32 -> [N][K] bf16
  transpose_cvt_kernel<<<dim3(3 * Cc / 32, Cc / 32),    256, 0, stream>>>(Wqkv, wqkv_t, Cc, 3 * Cc);
  transpose_cvt_kernel<<<dim3(Cc / 32,     Cc / 32),    256, 0, stream>>>(Wo,   wo_t,   Cc, Cc);
  transpose_cvt_kernel<<<dim3(INNERn / 32, Cc / 32),    256, 0, stream>>>(W1,   w1_t,   Cc, INNERn);
  transpose_cvt_kernel<<<dim3(INNERn / 32, Cc / 32),    256, 0, stream>>>(W2,   w2_t,   Cc, INNERn);
  transpose_cvt_kernel<<<dim3(Cc / 32,     INNERn / 32),256, 0, stream>>>(W3,   w3_t,   INNERn, Cc);

  // a = LN1(x) -> bf16
  ln_bf16_kernel<<<ROWS, 256, 0, stream>>>(x, ln1w, ln1b, a_bf);

  // qkv = a @ W_qkv   (4096 x 6144 x 2048)
  gemm_bf16_kernel<false><<<dim3(3 * Cc / GBN, ROWS / GBM), 256, 0, stream>>>(
      a_bf, wqkv_t, qkvf, nullptr, ROWS, 3 * Cc, Cc);

  // RoPE + split (V pre-transposed)
  rope_split_kernel<<<Bb * Tt * Hh, 64, 0, stream>>>(qkvf, cosT, sinT, qb, kb, vtb);

  // causal flash attention -> y bf16 [B,T,C]
  flash_attn_kernel<<<dim3(Bb * Hh, Tt / FA_BM), 128, 0, stream>>>(qb, kb, vtb, yb);

  // x2 = x + y @ W_o
  gemm_bf16_kernel<true><<<dim3(Cc / GBN, ROWS / GBM), 256, 0, stream>>>(
      yb, wo_t, x2, x, ROWS, Cc, Cc);

  // h = LN2(x2) -> bf16
  ln_bf16_kernel<<<ROWS, 256, 0, stream>>>(x2, ln2w, ln2b, hb);

  // g1 = h @ W1 ; g2 = h @ W2   (4096 x 8192 x 2048)
  gemm_bf16_kernel<false><<<dim3(INNERn / GBN, ROWS / GBM), 256, 0, stream>>>(
      hb, w1_t, g1, nullptr, ROWS, INNERn, Cc);
  gemm_bf16_kernel<false><<<dim3(INNERn / GBN, ROWS / GBM), 256, 0, stream>>>(
      hb, w2_t, g2, nullptr, ROWS, INNERn, Cc);

  // m = silu(g1) * g2 -> bf16
  swiglu_kernel<<<8192, 256, 0, stream>>>(g1, g2, mb, (long)ROWS * INNERn);

  // out = x2 + m @ W3   (4096 x 2048 x 8192)
  gemm_bf16_kernel<true><<<dim3(Cc / GBN, ROWS / GBM), 256, 0, stream>>>(
      mb, w3_t, out, x2, ROWS, Cc, INNERn);
}
